// MHA_6811818131832
// MI455X (gfx1250) — matrix operand order repaired and checked by reading, not compiled, not run on hardware
//
#include <hip/hip_runtime.h>

// MHA forward for MI455X (gfx1250, wave32, WMMA, async global->LDS DMA).
// Stages: f32->f16 convert, QKV GEMM (WMMA f16), weighted-flash attention
// (WMMA f16 logits + PV + ones-WMMA row sums), output GEMM (WMMA f16, f32 out).
// Global->LDS tile staging uses GLOBAL_LOAD_ASYNC_TO_LDS_B128 with LDS
// double-buffering (ASYNCcnt-tracked), overlapping DMA with WMMA compute.

#define Bq   4
#define Sq   1024
#define Dq   1024
#define Hq   16
#define HDq  64
#define NQKV 3072

typedef _Float16 half8 __attribute__((ext_vector_type(8)));
typedef _Float16 v16h  __attribute__((ext_vector_type(16)));
typedef float    v8f   __attribute__((ext_vector_type(8)));

__device__ __forceinline__ v8f v8f_zero() {
    v8f z;
    for (int i = 0; i < 8; ++i) z[i] = 0.0f;
    return z;
}

__device__ __forceinline__ v8f wmma_f16(v16h a, v16h b, v8f c) {
    // v_wmma_f32_16x16x32_f16: (neg_a, A, neg_b, B, c_mod, C, reuse_a, reuse_b)
    return __builtin_amdgcn_wmma_f32_16x16x32_f16(false, a, false, b, (short)0, c,
                                                  false, false);
}

// Flat pointers to __shared__ are {SHARED_BASE_hi32, lds_offset}; the low 32
// bits are the workgroup-relative LDS byte address used by DS/async ops.
__device__ __forceinline__ unsigned lds_off(const void* p) {
    return (unsigned)(unsigned long long)p;
}

// Async DMA: 16 bytes global -> LDS, no VGPR round trip, tracked by ASYNCcnt.
__device__ __forceinline__ void async_cp16(unsigned lds, const void* g) {
    asm volatile("global_load_async_to_lds_b128 %0, %1, off"
                 :: "v"(lds), "v"((unsigned long long)g)
                 : "memory");
}
__device__ __forceinline__ void async_wait0() {
    asm volatile("s_wait_asynccnt 0x0" ::: "memory");
}

// A-matrix 16x32 f16 fragment (ISA 7.12.2): lane l holds row (l&15);
// halves 0-7 = K 8*(l>>4)+0..7, halves 8-15 = K 8*(l>>4)+16..23.
__device__ __forceinline__ v16h ld_frag_a(const _Float16* base, int stride, int lane) {
    const _Float16* p = base + (size_t)(lane & 15) * stride + ((lane >> 4) << 3);
    half8 lo = *(const half8*)(p);
    half8 hi = *(const half8*)(p + 16);
    return __builtin_shufflevector(lo, hi, 0, 1, 2, 3, 4, 5, 6, 7,
                                   8, 9, 10, 11, 12, 13, 14, 15);
}

// B-matrix 32x16 f16 fragment: lane l holds column (l&15); halves 0-7 =
// K 8*(l>>4)+0..7, halves 8-15 = K 8*(l>>4)+16..23 (same as A). `base` points at
// a [N][K] row-major buffer (K contiguous per column of B).
__device__ __forceinline__ v16h ld_frag_b(const _Float16* base, int stride, int lane) {
    const _Float16* p = base + (size_t)(lane & 15) * stride + ((lane >> 4) << 3);
    half8 lo = *(const half8*)(p);
    half8 hi = *(const half8*)(p + 16);
    return __builtin_shufflevector(lo, hi, 0, 1, 2, 3, 4, 5, 6, 7,
                                   8, 9, 10, 11, 12, 13, 14, 15);
}

// ---------------------------------------------------------------- convert ---
__global__ __launch_bounds__(256) void cvt_f32_f16(const float* __restrict__ s,
                                                   _Float16* __restrict__ d, int n) {
    int i = (blockIdx.x * 256 + threadIdx.x) * 8;
    if (i >= n) return;
    float4 a = *(const float4*)(s + i);
    float4 b = *(const float4*)(s + i + 4);
    half8 h;
    h[0] = (_Float16)a.x; h[1] = (_Float16)a.y;
    h[2] = (_Float16)a.z; h[3] = (_Float16)a.w;
    h[4] = (_Float16)b.x; h[5] = (_Float16)b.y;
    h[6] = (_Float16)b.z; h[7] = (_Float16)b.w;
    *(half8*)(d + i) = h;
}

// ------------------------------------------------------------------- GEMM ---
// C[M,N] = A[M,K] * Bw[N,K]^T + bias.  Block tile 128x128, 8 waves, each wave
// 64x32 (4x2 WMMA tiles).  Tiles stream in via async DMA, double-buffered.
// MODE 0: write f32 to outF.  MODE 1: scatter f16 to Q/K [B,H,S,64] and
// V transposed [B,H,64,S], decoding the per-head interleaved rows of W_qkv.
template <int MODE>
__global__ __launch_bounds__(256) void gemm_f16_kernel(
    const _Float16* __restrict__ A, const _Float16* __restrict__ Bw,
    const float* __restrict__ bias, int K, int N,
    float* __restrict__ outF,
    _Float16* __restrict__ Qh, _Float16* __restrict__ Kh, _Float16* __restrict__ Vh) {
    __shared__ __align__(16) _Float16 Ash[2][128 * 48];
    __shared__ __align__(16) _Float16 Bsh[2][128 * 48];

    const int tid  = threadIdx.x;
    const int lane = tid & 31;
    const int w    = tid >> 5;
    const int wm   = (w >> 2) * 64;
    const int wn   = (w & 3) * 32;
    const int m0   = blockIdx.x * 128;
    const int n0   = blockIdx.y * 128;

    auto stage = [&](int bufi, int kc) {
        for (int i = 0; i < 2; ++i) {
            int idx = tid + i * 256;
            int row = idx >> 2;
            int seg = (idx & 3) * 8;
            async_cp16(lds_off(&Ash[bufi][row * 48 + seg]),
                       &A[(size_t)(m0 + row) * K + kc + seg]);
            async_cp16(lds_off(&Bsh[bufi][row * 48 + seg]),
                       &Bw[(size_t)(n0 + row) * K + kc + seg]);
        }
    };

    v8f acc[4][2];
    for (int tm = 0; tm < 4; ++tm)
        for (int tn = 0; tn < 2; ++tn) acc[tm][tn] = v8f_zero();

    stage(0, 0);
    async_wait0();
    __syncthreads();

    int buf = 0;
    for (int kc = 0; kc < K; kc += 32) {
        if (kc + 32 < K) stage(buf ^ 1, kc + 32);  // DMA next tile during compute

        v16h af[4], bf[2];
        for (int tm = 0; tm < 4; ++tm)
            af[tm] = ld_frag_a(&Ash[buf][(wm + tm * 16) * 48], 48, lane);
        for (int tn = 0; tn < 2; ++tn)
            bf[tn] = ld_frag_b(&Bsh[buf][(wn + tn * 16) * 48], 48, lane);
        for (int tm = 0; tm < 4; ++tm)
            for (int tn = 0; tn < 2; ++tn)
                acc[tm][tn] = wmma_f16(af[tm], bf[tn], acc[tm][tn]);

        async_wait0();
        __syncthreads();
        buf ^= 1;
    }

    // Rolled epilogue: recycle address temps per tile so accumulators stay
    // resident in VGPRs (avoid scratch spills).
    const int rhalf = 8 * (lane >> 4);
    const int ncol  = lane & 15;
#pragma unroll 1
    for (int tm = 0; tm < 4; ++tm) {
#pragma unroll 1
        for (int tn = 0; tn < 2; ++tn) {
            int n  = n0 + wn + tn * 16 + ncol;
            float bv = bias[n];
#pragma unroll 1
            for (int r = 0; r < 8; ++r) {
                int m   = m0 + wm + tm * 16 + r + rhalf;
                float v = acc[tm][tn][r] + bv;
                if (MODE == 0) {
                    outF[(size_t)m * N + n] = v;
                } else {
                    int b = m >> 10, s = m & 1023;
                    int h = n / 192, rr = n % 192;
                    _Float16 hv = (_Float16)v;
                    if (rr < 64) {
                        Qh[(((size_t)b * Hq + h) * Sq + s) * HDq + rr] = hv;
                    } else if (rr < 128) {
                        Kh[(((size_t)b * Hq + h) * Sq + s) * HDq + rr - 64] = hv;
                    } else {  // V stored transposed: [B,H,HD,S]
                        Vh[(((size_t)b * Hq + h) * HDq + (rr - 128)) * Sq + s] = hv;
                    }
                }
            }
        }
    }
}

// -------------------------------------------------------------- attention ---
// grid (B*H, S/128), 256 threads. Wave w owns 16 q-rows; K (row-major) and
// V^T chunks stream in via async DMA, double-buffered.  Online weighted
// softmax: m = max(logits); e = w*exp(logits-m); O = sum e*V; out = O/sum(e).
// Row sums of the probability tile come from an extra WMMA against an
// all-ones B fragment (C-layout rows match the lrow[] register layout).
__global__ __launch_bounds__(256) void attn_kernel(
    const _Float16* __restrict__ Qh, const _Float16* __restrict__ Kh,
    const _Float16* __restrict__ Vh, const int* __restrict__ mask,
    const float* __restrict__ wts, _Float16* __restrict__ valsh) {
    __shared__ __align__(16) _Float16 Klds[2][32 * 80];   // [kcol][hd]
    __shared__ __align__(16) _Float16 Vt[2][64 * 40];     // [hd][kcol]
    __shared__ __align__(16) _Float16 Pw[8 * 16 * 40];    // per-wave P scratch

    const int tid  = threadIdx.x;
    const int lane = tid & 31;
    const int w    = tid >> 5;
    const int bh   = blockIdx.x;
    const int b    = bh >> 4;
    const int h    = bh & 15;
    const int q0   = blockIdx.y * 128 + w * 16;
    const int rhalf = 8 * (lane >> 4);
    const int ncol  = lane & 15;

    auto stageKV = [&](int bufi, int kc) {
        {   // K chunk: 32 rows x 64 halves (128B = 8 chunks/row)
            int row = tid >> 3;
            int c8  = (tid & 7) * 8;
            async_cp16(lds_off(&Klds[bufi][row * 80 + c8]),
                       Kh + (((size_t)bh) * Sq + kc + row) * HDq + c8);
        }
        {   // V^T chunk: 64 rows(hd) x 32 halves(kcol) (4 chunks/row)
            int row = tid >> 2;
            int c8  = (tid & 3) * 8;
            async_cp16(lds_off(&Vt[bufi][row * 40 + c8]),
                       Vh + (((size_t)bh) * HDq + row) * Sq + kc + c8);
        }
    };

    float mrow[8], lrow[8];
    v8f o[4];
#pragma unroll
    for (int r = 0; r < 8; ++r) { mrow[r] = -__FLT_MAX__; lrow[r] = 0.0f; }
    for (int t = 0; t < 4; ++t) o[t] = v8f_zero();

    v16h ones;  // all-ones B fragment: P x ones -> per-row sums of P
#pragma unroll
    for (int i = 0; i < 16; ++i) ones[i] = (_Float16)1.0f;

    const _Float16* qbase = Qh + (((size_t)bh) * Sq + q0) * HDq;
    v16h aq0 = ld_frag_a(qbase + 0, HDq, lane);   // hd 0..31
    v16h aq1 = ld_frag_a(qbase + 32, HDq, lane);  // hd 32..63
    _Float16* pw = &Pw[w * 16 * 40];

    stageKV(0, 0);
    async_wait0();
    __syncthreads();

    int buf = 0;
    for (int kc = 0; kc < Sq; kc += 32) {
        if (kc + 32 < Sq) stageKV(buf ^ 1, kc + 32);  // DMA next K/V chunk

        // logits: 16 q-rows x 32 k-cols, 2 n-tiles x 2 K-steps
        v8f st[2];
#pragma unroll
        for (int nt = 0; nt < 2; ++nt) {
            v8f sacc = v8f_zero();
            v16h b0 = ld_frag_b(&Klds[buf][(nt * 16) * 80 + 0], 80, lane);
            v16h b1 = ld_frag_b(&Klds[buf][(nt * 16) * 80 + 32], 80, lane);
            sacc = wmma_f16(aq0, b0, sacc);
            sacc = wmma_f16(aq1, b1, sacc);
            st[nt] = sacc;
        }

        float alpha[8];
#pragma unroll
        for (int r = 0; r < 8; ++r) {
            int qrow = q0 + r + rhalf;
            int k0   = kc + ncol;
            int k1   = kc + 16 + ncol;
            float l0 = st[0][r] * 0.125f;   // 1/sqrt(64)
            float l1 = st[1][r] * 0.125f;
            if (mask[((size_t)b * Sq + qrow) * Sq + k0] == 0) l0 = -9.0e15f;
            if (mask[((size_t)b * Sq + qrow) * Sq + k1] == 0) l1 = -9.0e15f;
            float mx = fmaxf(l0, l1);
            mx = fmaxf(mx, __shfl_xor(mx, 1));
            mx = fmaxf(mx, __shfl_xor(mx, 2));
            mx = fmaxf(mx, __shfl_xor(mx, 4));
            mx = fmaxf(mx, __shfl_xor(mx, 8));
            float nm = fmaxf(mrow[r], mx);
            alpha[r] = __expf(mrow[r] - nm);
            mrow[r]  = nm;
            float w0 = wts[((size_t)b * Sq + k0) * Sq + qrow];  // transposed weights
            float w1 = wts[((size_t)b * Sq + k1) * Sq + qrow];
            float p0 = w0 * __expf(l0 - nm);
            float p1 = w1 * __expf(l1 - nm);
            pw[(r + rhalf) * 40 + ncol]      = (_Float16)p0;
            pw[(r + rhalf) * 40 + 16 + ncol] = (_Float16)p1;
        }

#pragma unroll
        for (int t = 0; t < 4; ++t)
#pragma unroll
            for (int r = 0; r < 8; ++r) o[t][r] *= alpha[r];

        // P (f16, via per-wave LDS) -> A fragment; row sums via ones-WMMA;
        // then O += P(16x32) x V(32x64).
        v16h ap = ld_frag_a(pw, 40, lane);
        v8f psum = wmma_f16(ap, ones, v8f_zero());
#pragma unroll
        for (int r = 0; r < 8; ++r) lrow[r] = lrow[r] * alpha[r] + psum[r];
#pragma unroll
        for (int t = 0; t < 4; ++t) {
            v16h bv = ld_frag_b(&Vt[buf][(t * 16) * 40], 40, lane);
            o[t] = wmma_f16(ap, bv, o[t]);
        }

        async_wait0();
        __syncthreads();
        buf ^= 1;
    }

#pragma unroll 1
    for (int r = 0; r < 8; ++r) {
        int qrow  = q0 + r + rhalf;
        float inv = 1.0f / lrow[r];
        for (int t = 0; t < 4; ++t)
            valsh[((size_t)b * Sq + qrow) * Dq + h * HDq + t * 16 + ncol] =
                (_Float16)(o[t][r] * inv);
    }
}

// ----------------------------------------------------------------- launch ---
extern "C" void kernel_launch(void* const* d_in, const int* in_sizes, int n_in,
                              void* d_out, int out_size, void* d_ws, size_t ws_size,
                              hipStream_t stream) {
    (void)in_sizes; (void)n_in; (void)out_size; (void)ws_size;
    const float* x    = (const float*)d_in[0];
    const int*   mask = (const int*)d_in[1];
    const float* wts  = (const float*)d_in[2];
    const float* Wqkv = (const float*)d_in[3];
    const float* bqkv = (const float*)d_in[4];
    const float* Wo   = (const float*)d_in[5];
    const float* bo   = (const float*)d_in[6];
    float* out = (float*)d_out;

    const size_t nX    = (size_t)Bq * Sq * Dq;        // 4M
    const size_t nWqkv = (size_t)NQKV * Dq;           // 3M
    const size_t nWo   = (size_t)Dq * Dq;             // 1M
    const size_t nHead = (size_t)Bq * Hq * Sq * HDq;  // 4M

    _Float16* xh    = (_Float16*)d_ws;
    _Float16* wqkvh = xh + nX;
    _Float16* woh   = wqkvh + nWqkv;
    _Float16* Qh    = woh + nWo;
    _Float16* Kh    = Qh + nHead;
    _Float16* Vh    = Kh + nHead;      // transposed per-head: [B,H,HD,S]
    _Float16* valsh = Vh + nHead;      // total ~50 MB of workspace

    cvt_f32_f16<<<(int)(nX / 2048), 256, 0, stream>>>(x, xh, (int)nX);
    cvt_f32_f16<<<(int)(nWqkv / 2048), 256, 0, stream>>>(Wqkv, wqkvh, (int)nWqkv);
    cvt_f32_f16<<<(int)(nWo / 2048), 256, 0, stream>>>(Wo, woh, (int)nWo);

    // QKV projection: [4096,1024] x [3072,1024]^T -> scatter Q/K/V
    gemm_f16_kernel<1><<<dim3(32, 24), 256, 0, stream>>>(
        xh, wqkvh, bqkv, Dq, NQKV, nullptr, Qh, Kh, Vh);

    // weighted-softmax attention
    attn_kernel<<<dim3(Bq * Hq, Sq / 128), 256, 0, stream>>>(
        Qh, Kh, Vh, mask, wts, valsh);

    // output projection: [4096,1024] x [1024,1024]^T + b_o -> f32 out
    gemm_f16_kernel<0><<<dim3(32, 8), 256, 0, stream>>>(
        valsh, woh, bo, Dq, Dq, out, nullptr, nullptr, nullptr);
}
